// MultiHeadAttention_38903813767539
// MI455X (gfx1250) — matrix operand order repaired and checked by reading, not compiled, not run on hardware
//
#include <hip/hip_runtime.h>
#include <hip/hip_bf16.h>

// ---------------------------------------------------------------------------
// MHA for MI455X (gfx1250): bf16 WMMA GEMMs + flash-style fused attention.
// B=2, S=2048, D=1024, H=16, d_k=64.
//  - All GEMMs on v_wmma_f32_16x16x32_bf16 (f32 accum).
//  - Shared B-operands (W tile / K,V tile per head) staged into LDS with
//    double-buffered GLOBAL_LOAD_ASYNC_TO_LDS_B128 (ASYNCcnt pipeline),
//    removing the 8x redundant global fetch across the 8 waves of a block.
//  - Scores never touch HBM (flash online-softmax), so the kernel is
//    matrix-core bound, not HBM bound.
// ---------------------------------------------------------------------------

typedef __attribute__((ext_vector_type(16))) __bf16 v16bf;
typedef __attribute__((ext_vector_type(8)))  float  v8f;

constexpr int Bc  = 2;
constexpr int Sc  = 2048;
constexpr int Dc  = 1024;
constexpr int Hc  = 16;
constexpr int DKc = 64;
constexpr int NTOK = Bc * Sc;   // 4096

// A-fragment K map for 16-bit 16x32 A (contiguous, matches B fills):
//   lanes 0-15:  V0..7 -> K=0..15
//   lanes 16-31: V0..7 -> K=16..31
__device__ __forceinline__ int kmapA(int j, int half) {
    return half * 16 + 2 * j;
}

union AF {
    v16bf v;
    unsigned int u[8];
    __bf16 b[16];
};

// --- CDNA5 async copy: global -> LDS, 16B per lane, tracked by ASYNCcnt ----
// Flat-aperture rule: low 32 bits of a generic shared pointer == LDS offset.
__device__ __forceinline__ void async_b128(const void* g, const void* l) {
    asm volatile("global_load_async_to_lds_b128 %0, %1, off"
                 :: "v"((unsigned)(uintptr_t)l),
                    "v"((unsigned long long)(uintptr_t)g)
                 : "memory");
}
#define WAIT_ASYNC(n) asm volatile("s_wait_asynccnt " #n ::: "memory")

// ---------------------------------------------------------------------------
// GEMM: Y = X @ W + bias. Per-wave 16x64 output tile; W tile staged in LDS
// via async copy, double buffered.
// MODE 0: X f32 [NTOK,D] -> bf16 out head-major [B,H,S,DK]
// MODE 1: X bf16 [NTOK,D] -> f32 out row-major [NTOK,D]
// ---------------------------------------------------------------------------
template<int MODE>
__global__ __launch_bounds__(256)
void proj_gemm(const float*  __restrict__ Xf,
               const __bf16* __restrict__ Xb,
               const float*  __restrict__ W,
               const float*  __restrict__ bias,
               __bf16* __restrict__ outB,
               float*  __restrict__ outF)
{
    __shared__ float Wt[2][32 * 64];   // 2 x 8 KB f32 K-tiles of W

    const int tid  = threadIdx.x;
    const int lane = tid & 31;
    const int wave = tid >> 5;
    const int half = lane >> 4;
    const int lc   = lane & 15;
    const int rowBase = blockIdx.y * 128 + wave * 16;
    const int colBase = blockIdx.x * 64;
    const int arow    = rowBase + lc;

    // stage a 32(K) x 64(N) f32 W tile: 512 x 16B segments, 2 per thread
    auto stage = [&](int k0, int buf) {
#pragma unroll
        for (int i = 0; i < 2; ++i) {
            const int seg = tid + i * 256;
            const int row = seg >> 4;
            const int c4  = (seg & 15) * 4;
            async_b128(W + (size_t)(k0 + row) * Dc + colBase + c4,
                       &Wt[buf][row * 64 + c4]);
        }
    };

    stage(0, 0);

    v8f acc[4] = {};
    for (int k0 = 0, it = 0; k0 < Dc; k0 += 32, ++it) {
        const int buf = it & 1;
        const bool more = (k0 + 32) < Dc;
        if (more) stage(k0 + 32, buf ^ 1);       // prefetch next tile
        if (more) WAIT_ASYNC(2); else WAIT_ASYNC(0);  // my 2 loads of tile `it` done
        __syncthreads();                          // everyone's writes visible

        // ---- A fragment: 16 rows x 32 K (per-wave, direct from global) ----
        AF a;
        if (MODE == 0) {
#pragma unroll
            for (int j = 0; j < 8; ++j) {
                const int k = k0 + kmapA(j, half);
                const float* p = Xf + (size_t)arow * Dc + k;
                a.b[2 * j]     = (__bf16)p[0];
                a.b[2 * j + 1] = (__bf16)p[1];
            }
        } else {
#pragma unroll
            for (int j = 0; j < 8; ++j) {
                const int k = k0 + kmapA(j, half);
                a.u[j] = *reinterpret_cast<const unsigned*>(Xb + (size_t)arow * Dc + k);
            }
        }

        // ---- 4 B fragments from LDS tile + 4 WMMAs (A reused) ----
        const float* Wb = Wt[buf];
#pragma unroll
        for (int t = 0; t < 4; ++t) {
            const int col = t * 16 + lc;
            AF bfr;
#pragma unroll
            for (int j = 0; j < 8; ++j) {
                const int k = half * 16 + 2 * j;
                bfr.b[2 * j]     = (__bf16)Wb[k * 64 + col];
                bfr.b[2 * j + 1] = (__bf16)Wb[(k + 1) * 64 + col];
            }
            acc[t] = __builtin_amdgcn_wmma_f32_16x16x32_bf16(
                false, a.v, false, bfr.v, (short)0, acc[t], false, false);
        }
        __syncthreads();                          // done reading tile `it`
    }

    // ---- bias + store ----
#pragma unroll
    for (int t = 0; t < 4; ++t) {
        const int col = colBase + t * 16 + lc;
        const float bb = bias[col];
#pragma unroll
        for (int i = 0; i < 8; ++i) {
            const float val = acc[t][i] + bb;
            const int row = rowBase + i + half * 8;
            if (MODE == 0) {
                const int b = row / Sc, s = row % Sc;
                const int h = col / DKc, dk = col % DKc;
                outB[(((size_t)(b * Hc + h)) * Sc + s) * DKc + dk] = (__bf16)val;
            } else {
                outF[(size_t)row * Dc + col] = val;
            }
        }
    }
}

// ---------------------------------------------------------------------------
// Flash attention. Block = 8 waves x 16 queries of one (b,h); keys streamed
// in chunks of 64, K/V chunk staged in LDS (async, double buffered) and
// shared by all waves. Masking is faithful to the reference: masked scores
// become 1e-10 (NOT -inf) and still feed the softmax denominator.
// ---------------------------------------------------------------------------
__global__ __launch_bounds__(256)
void attn_kernel(const __bf16* __restrict__ Qb,
                 const __bf16* __restrict__ Kb,
                 const __bf16* __restrict__ Vb,
                 const unsigned char* __restrict__ pad,   // [B, S] bool
                 __bf16* __restrict__ Ob)                 // [B, S, H, DK]
{
    __shared__ __bf16 Kt[2][64 * 64];     // 2 x 8 KB key tiles
    __shared__ __bf16 Vt[2][64 * 64];     // 2 x 8 KB value tiles
    __shared__ __bf16 Pt[8 * 16 * 64];    // per-wave P tiles (16 KB)

    const int tid  = threadIdx.x;
    const int lane = tid & 31;
    const int wave = tid >> 5;
    const int half = lane >> 4;
    const int lc   = lane & 15;
    const int bh   = blockIdx.y;          // b*H + h
    const int b    = bh / Hc;
    const int h    = bh % Hc;
    const int qBase = blockIdx.x * 128 + wave * 16;

    const size_t headOff = (size_t)bh * Sc * DKc;
    const __bf16* Qh = Qb + headOff;
    const __bf16* Kh = Kb + headOff;
    const __bf16* Vh = Vb + headOff;

    // stage 64x64 bf16 K and V chunks: 512 x 16B segs each, 2+2 per thread
    auto stage = [&](int kc, int buf) {
#pragma unroll
        for (int i = 0; i < 2; ++i) {
            const int seg = tid + i * 256;
            const int row = seg >> 3;             // 64 keys
            const int c8  = (seg & 7) * 8;        // 8 bf16 per seg
            async_b128(Kh + (size_t)(kc + row) * DKc + c8, &Kt[buf][row * 64 + c8]);
            async_b128(Vh + (size_t)(kc + row) * DKc + c8, &Vt[buf][row * 64 + c8]);
        }
    };

    // ---- Q fragments held in registers for the whole key loop ----
    AF aq0, aq1;
    {
        const int r = qBase + lc;
#pragma unroll
        for (int j = 0; j < 8; ++j) {
            const int k = kmapA(j, half);
            aq0.u[j] = *reinterpret_cast<const unsigned*>(Qh + (size_t)r * DKc + k);
            aq1.u[j] = *reinterpret_cast<const unsigned*>(Qh + (size_t)r * DKc + 32 + k);
        }
    }

    float m[8], l[8];
    v8f o[4] = {};
#pragma unroll
    for (int i = 0; i < 8; ++i) { m[i] = -__builtin_huge_valf(); l[i] = 0.f; }

    __bf16* myP = Pt + wave * 16 * 64;

    stage(0, 0);

    for (int kc = 0, it = 0; kc < Sc; kc += 64, ++it) {
        const int buf = it & 1;
        const bool more = (kc + 64) < Sc;
        if (more) stage(kc + 64, buf ^ 1);       // prefetch next K/V chunk
        if (more) WAIT_ASYNC(4); else WAIT_ASYNC(0);  // my 4 loads of chunk `it` done
        __syncthreads();

        // ---------------- scores: 4 tiles of 16 keys ----------------
        v8f sc[4];
#pragma unroll
        for (int t = 0; t < 4; ++t) {
            const int kl  = t * 16 + lc;          // key within chunk
            const int key = kc + kl;
            AF bk0, bk1;
#pragma unroll
            for (int j = 0; j < 8; ++j) {
                const int kd = half * 16 + 2 * j;
                bk0.u[j] = *reinterpret_cast<const unsigned*>(&Kt[buf][kl * 64 + kd]);
                bk1.u[j] = *reinterpret_cast<const unsigned*>(&Kt[buf][kl * 64 + 32 + kd]);
            }
            v8f c = {};
            c = __builtin_amdgcn_wmma_f32_16x16x32_bf16(false, aq0.v, false, bk0.v,
                                                        (short)0, c, false, false);
            c = __builtin_amdgcn_wmma_f32_16x16x32_bf16(false, aq1.v, false, bk1.v,
                                                        (short)0, c, false, false);
            const bool padc = pad[(size_t)b * Sc + key] != 0;
#pragma unroll
            for (int i = 0; i < 8; ++i) {
                float s = c[i] * 0.125f;               // 1/sqrt(64)
                const int row = qBase + i + half * 8;
                if (key > row && s != 0.f) s = 1e-10f; // triu(scores,1) != 0
                if (padc)                  s = 1e-10f; // padding mask
                c[i] = s;
            }
            sc[t] = c;
        }

        // ---------------- online softmax update ----------------
        float alpha[8], rs[8];
#pragma unroll
        for (int i = 0; i < 8; ++i) {
            float v = fmaxf(fmaxf(sc[0][i], sc[1][i]), fmaxf(sc[2][i], sc[3][i]));
            v = fmaxf(v, __shfl_xor(v, 1, 16));
            v = fmaxf(v, __shfl_xor(v, 2, 16));
            v = fmaxf(v, __shfl_xor(v, 4, 16));
            v = fmaxf(v, __shfl_xor(v, 8, 16));
            const float mn = fmaxf(m[i], v);
            alpha[i] = __expf(m[i] - mn);
            m[i] = mn;
            rs[i] = 0.f;
        }
#pragma unroll
        for (int t = 0; t < 4; ++t) {
#pragma unroll
            for (int i = 0; i < 8; ++i) {
                const float p = __expf(sc[t][i] - m[i]);
                rs[i] += p;
                myP[(i + half * 8) * 64 + t * 16 + lc] = (__bf16)p;
            }
        }
#pragma unroll
        for (int i = 0; i < 8; ++i) {
            float v = rs[i];
            v += __shfl_xor(v, 1, 16);
            v += __shfl_xor(v, 2, 16);
            v += __shfl_xor(v, 4, 16);
            v += __shfl_xor(v, 8, 16);
            l[i] = l[i] * alpha[i] + v;
        }
#pragma unroll
        for (int t = 0; t < 4; ++t)
#pragma unroll
            for (int i = 0; i < 8; ++i)
                o[t][i] *= alpha[i];

        // per-wave P store -> load fence (DS ops are in order per wave)
        asm volatile("s_wait_dscnt 0x0" ::: "memory");

        // ---------------- O += P(16x64) @ V(64x64) ----------------
#pragma unroll
        for (int kk = 0; kk < 64; kk += 32) {
            AF ap;
#pragma unroll
            for (int j = 0; j < 8; ++j) {
                const int k = kk + kmapA(j, half);
                ap.u[j] = *reinterpret_cast<const unsigned*>(myP + lc * 64 + k);
            }
#pragma unroll
            for (int t = 0; t < 4; ++t) {
                const int dk = t * 16 + lc;
                AF bv;
#pragma unroll
                for (int j = 0; j < 8; ++j) {
                    const int kl = kk + half * 16 + 2 * j;
                    bv.b[2 * j]     = Vt[buf][kl * 64 + dk];
                    bv.b[2 * j + 1] = Vt[buf][(kl + 1) * 64 + dk];
                }
                o[t] = __builtin_amdgcn_wmma_f32_16x16x32_bf16(
                    false, ap.v, false, bv.v, (short)0, o[t], false, false);
            }
        }
        __syncthreads();   // all waves done reading chunk `it` before overwrite
    }

    // ---- normalize + store (token-major [B,S,H,DK] = concat layout) ----
#pragma unroll
    for (int t = 0; t < 4; ++t) {
        const int dk = t * 16 + lc;
#pragma unroll
        for (int i = 0; i < 8; ++i) {
            const int q = qBase + i + half * 8;
            const float val = o[t][i] / l[i];
            Ob[(((size_t)b * Sc + q) * Hc + h) * DKc + dk] = (__bf16)val;
        }
    }
}

// ---------------------------------------------------------------------------
extern "C" void kernel_launch(void* const* d_in, const int* in_sizes, int n_in,
                              void* d_out, int out_size, void* d_ws, size_t ws_size,
                              hipStream_t stream)
{
    const float* q  = (const float*)d_in[0];
    const float* k  = (const float*)d_in[1];
    const float* v  = (const float*)d_in[2];
    const unsigned char* pad = (const unsigned char*)d_in[3];
    const float* Wq = (const float*)d_in[4];
    const float* bq = (const float*)d_in[5];
    const float* Wk = (const float*)d_in[6];
    const float* bk = (const float*)d_in[7];
    const float* Wv = (const float*)d_in[8];
    const float* bv = (const float*)d_in[9];
    const float* Wo = (const float*)d_in[10];
    const float* bo = (const float*)d_in[11];
    float* out = (float*)d_out;

    // workspace: bf16 Q,K,V (head-major) + bf16 attention output = 32 MB
    const size_t per = (size_t)Bc * Sc * Dc;
    __bf16* qb = (__bf16*)d_ws;
    __bf16* kb = qb + per;
    __bf16* vb = kb + per;
    __bf16* ab = vb + per;

    dim3 blk(256);
    dim3 gridG(Dc / 64, NTOK / 128);     // (16, 32)
    proj_gemm<0><<<gridG, blk, 0, stream>>>(q, nullptr, Wq, bq, qb, nullptr);
    proj_gemm<0><<<gridG, blk, 0, stream>>>(k, nullptr, Wk, bk, kb, nullptr);
    proj_gemm<0><<<gridG, blk, 0, stream>>>(v, nullptr, Wv, bv, vb, nullptr);

    dim3 gridA(Sc / 128, Bc * Hc);       // (16, 32)
    attn_kernel<<<gridA, blk, 0, stream>>>(qb, kb, vb, pad, ab);

    proj_gemm<1><<<gridG, blk, 0, stream>>>(nullptr, ab, Wo, bo, nullptr, out);
}
